// _NonLocalBlockND_3100966388111
// MI455X (gfx1250) — matrix operand order repaired and checked by reading, not compiled, not run on hardware
//
#include <hip/hip_runtime.h>
#include <hip/hip_bf16.h>

// ---------------------------------------------------------------------------
// NonLocalBlockND (B=4, C=1024, T=8, H=32, W=32, CI=512) for gfx1250 (CDNA5)
// All GEMMs run on v_wmma_f32_16x16x32_bf16 (wave32 WMMA).
// Wave tile 64x64 (16 WMMAs / 8 fragment loads per k-step), block tile 128x128.
// ---------------------------------------------------------------------------

typedef __bf16 bf16;
typedef __attribute__((ext_vector_type(16))) __bf16 v16bf;
typedef __attribute__((ext_vector_type(8)))  __bf16 v8bf;
typedef __attribute__((ext_vector_type(8)))  float  v8f;

static constexpr int Bn  = 4;
static constexpr int C   = 1024;
static constexpr int CI  = 512;
static constexpr int T   = 8;
static constexpr int H   = 32;
static constexpr int W   = 32;
static constexpr int N   = T * H * W;             // 8192 positions
static constexpr int Mp  = T * (H / 2) * (W / 2); // 2048 pooled positions
static constexpr float BN_EPS = 1e-5f;

__device__ __forceinline__ bf16 to_bf16(float f) { return (bf16)f; }

// ----- WMMA fragment loaders (NT: both operands row-major over K) ----------
// A-matrix 16x32 bf16 layout (ISA 7.12.2): lane r (half h) holds row M=r,
// elements [0..7] = K = k0+8h .. k0+8h+7, elements [8..15] = K = k0+16+8h ..
__device__ __forceinline__ v16bf load_frag_a(const bf16* __restrict__ base,
                                             int row, int ldk, int k0, int lane) {
    int r = lane & 15, h = (lane >> 4) & 1;
    const bf16* p = base + (size_t)(row + r) * ldk + k0 + h * 8;
    union { v16bf v; v8bf q[2]; } u;
    u.q[0] = *(const v8bf*)(p);
    u.q[1] = *(const v8bf*)(p + 16);
    return u.v;
}

// B-matrix 32x16 bf16 layout (ISA 7.12.4 pattern): lane r (half h) holds
// column N=r, elements [0..7] = K = k0+8h .. k0+8h+7, elements [8..15] = K = k0+16+8h ..
__device__ __forceinline__ v16bf load_frag_b(const bf16* __restrict__ base,
                                             int col, int ldk, int k0, int lane) {
    int r = lane & 15, h = (lane >> 4) & 1;
    const bf16* p = base + (size_t)(col + r) * ldk + k0 + h * 8;
    union { v16bf v; v8bf q[2]; } u;
    u.q[0] = *(const v8bf*)(p);
    u.q[1] = *(const v8bf*)(p + 16);
    return u.v;
}

// ----- generic NT GEMM, bf16 in, bf16 out --------------------------------
// Out = scale * (A(Mo x K) * B(Nn x K)^T) + bias[m-dim]
// transposed==0: Out[m * Nn + n]   (row-major Mo x Nn, scattered stores)
// transposed==1: Out[n * Mo + m]   (row-major Nn x Mo, contiguous v8bf stores)
__global__ __launch_bounds__(128)
void gemm_nt_bf16(const bf16* __restrict__ A, const bf16* __restrict__ Bm,
                  bf16* __restrict__ Out, const float* __restrict__ bias,
                  float scale, int transposed,
                  int Mo, int Nn, int K,
                  size_t aBatch, size_t bBatch, size_t oBatch) {
    const int b    = blockIdx.z;
    const int lane = threadIdx.x & 31;
    const int wave = threadIdx.x >> 5;               // 0..3 (2x2)
    const int mBase = blockIdx.y * 128 + (wave >> 1) * 64;
    const int nBase = blockIdx.x * 128 + (wave & 1)  * 64;

    const bf16* Ab = A  + (size_t)b * aBatch;
    const bf16* Bb = Bm + (size_t)b * bBatch;
    bf16*       Ob = Out + (size_t)b * oBatch;

    v8f acc[4][4];
    #pragma unroll
    for (int i = 0; i < 4; ++i)
        #pragma unroll
        for (int j = 0; j < 4; ++j)
            acc[i][j] = (v8f){};

    #pragma unroll 1
    for (int k0 = 0; k0 < K; k0 += 32) {
        v16bf af[4], bf[4];
        #pragma unroll
        for (int i = 0; i < 4; ++i)
            af[i] = load_frag_a(Ab, mBase + 16 * i, K, k0, lane);
        #pragma unroll
        for (int j = 0; j < 4; ++j)
            bf[j] = load_frag_b(Bb, nBase + 16 * j, K, k0, lane);
        #pragma unroll
        for (int i = 0; i < 4; ++i)
            #pragma unroll
            for (int j = 0; j < 4; ++j)
                acc[i][j] = __builtin_amdgcn_wmma_f32_16x16x32_bf16(
                    false, af[i], false, bf[j], (short)0, acc[i][j], false, false);
    }

    const int r = lane & 15, h = (lane >> 4) & 1;
    #pragma unroll
    for (int i = 0; i < 4; ++i) {
        #pragma unroll
        for (int j = 0; j < 4; ++j) {
            const int mT = mBase + i * 16;
            const int nT = nBase + j * 16;
            const v8f& a = acc[i][j];
            if (transposed) {
                const int row = nT + r;
                const int col = mT + h * 8;
                v8bf pk;
                #pragma unroll
                for (int e = 0; e < 8; ++e) {
                    float v = a[e] * scale + (bias ? bias[col + e] : 0.0f);
                    pk[e] = to_bf16(v);
                }
                *(v8bf*)(Ob + (size_t)row * Mo + col) = pk;
            } else {
                const int n = nT + r;
                const int m = mT + h * 8;
                #pragma unroll
                for (int e = 0; e < 8; ++e) {
                    float v = a[e] * scale + (bias ? bias[m + e] : 0.0f);
                    Ob[(size_t)(m + e) * Nn + n] = to_bf16(v);
                }
            }
        }
    }
}

// ----- final GEMM: w_y = w_w * yT^T + w_b; BN; + x; fp32 out ---------------
__global__ __launch_bounds__(128)
void gemm_final(const bf16* __restrict__ A,      // w_w bf16 (C x CI)
                const bf16* __restrict__ Bm,     // yT (N x CI) per batch
                const float* __restrict__ wb,
                const float* __restrict__ gamma, const float* __restrict__ beta,
                const float* __restrict__ mean,  const float* __restrict__ var,
                const float* __restrict__ x, float* __restrict__ out,
                int Mo, int Nn, int K, size_t bBatch, size_t xBatch) {
    const int b    = blockIdx.z;
    const int lane = threadIdx.x & 31;
    const int wave = threadIdx.x >> 5;
    const int mBase = blockIdx.y * 128 + (wave >> 1) * 64;
    const int nBase = blockIdx.x * 128 + (wave & 1)  * 64;

    const bf16* Bb = Bm + (size_t)b * bBatch;
    const float* xb = x + (size_t)b * xBatch;
    float* ob = out + (size_t)b * xBatch;

    v8f acc[4][4];
    #pragma unroll
    for (int i = 0; i < 4; ++i)
        #pragma unroll
        for (int j = 0; j < 4; ++j)
            acc[i][j] = (v8f){};

    #pragma unroll 1
    for (int k0 = 0; k0 < K; k0 += 32) {
        v16bf af[4], bf[4];
        #pragma unroll
        for (int i = 0; i < 4; ++i)
            af[i] = load_frag_a(A, mBase + 16 * i, K, k0, lane);
        #pragma unroll
        for (int j = 0; j < 4; ++j)
            bf[j] = load_frag_b(Bb, nBase + 16 * j, K, k0, lane);
        #pragma unroll
        for (int i = 0; i < 4; ++i)
            #pragma unroll
            for (int j = 0; j < 4; ++j)
                acc[i][j] = __builtin_amdgcn_wmma_f32_16x16x32_bf16(
                    false, af[i], false, bf[j], (short)0, acc[i][j], false, false);
    }

    const int r = lane & 15, h = (lane >> 4) & 1;
    #pragma unroll
    for (int i = 0; i < 4; ++i) {
        #pragma unroll
        for (int j = 0; j < 4; ++j) {
            const int mT = mBase + i * 16;
            const int nT = nBase + j * 16;
            const v8f& a = acc[i][j];
            const int n = nT + r;
            const int m = mT + h * 8;
            #pragma unroll
            for (int e = 0; e < 8; ++e) {
                const int ch = m + e;
                float inv = gamma[ch] * rsqrtf(var[ch] + BN_EPS);
                float v = a[e] + wb[ch];
                v = (v - mean[ch]) * inv + beta[ch];
                size_t idx = (size_t)ch * Nn + n;
                ob[idx] = v + xb[idx];
            }
        }
    }
}

// ----- x (B,C,N) f32 -> xT (B,N,C) bf16, LDS tile transpose ----------------
__global__ __launch_bounds__(256)
void transpose_f32_to_bf16(const float* __restrict__ X, bf16* __restrict__ XT) {
    __shared__ bf16 tile[32][33];
    const int b  = blockIdx.z;
    const int c0 = blockIdx.y * 32;
    const int n0 = blockIdx.x * 32;
    const float* Xb  = X  + (size_t)b * C * N;
    bf16*        XTb = XT + (size_t)b * N * C;
    for (int i = threadIdx.y; i < 32; i += 8)
        tile[i][threadIdx.x] = to_bf16(Xb[(size_t)(c0 + i) * N + n0 + threadIdx.x]);
    __syncthreads();
    for (int i = threadIdx.y; i < 32; i += 8)
        XTb[(size_t)(n0 + i) * C + c0 + threadIdx.x] = tile[threadIdx.x][i];
}

// ----- fp32 -> bf16 elementwise (weights) ----------------------------------
__global__ __launch_bounds__(256)
void f32_to_bf16_kernel(const float* __restrict__ in, bf16* __restrict__ out, int n) {
    int i = blockIdx.x * 256 + threadIdx.x;
    if (i < n) out[i] = to_bf16(in[i]);
}

// ----- MaxPool (1,2,2) over (B,CI,T,H,W) bf16 -> (B,CI,T,H/2,W/2) bf16 -----
__global__ __launch_bounds__(256)
void pool_hw(const bf16* __restrict__ In, bf16* __restrict__ Out) {
    size_t i = (size_t)blockIdx.x * 256 + threadIdx.x;   // over B*CI*Mp = 2^22
    const int m  = (int)(i & (Mp - 1));
    const size_t bc = i >> 11;                           // b*CI + c
    const int t  = m >> 8;
    const int h2 = (m >> 4) & 15;
    const int w2 = m & 15;
    const bf16* p = In + bc * (size_t)N + (size_t)t * (H * W) + (size_t)(2 * h2) * W + 2 * w2;
    float v0 = (float)p[0], v1 = (float)p[1];
    float v2 = (float)p[W], v3 = (float)p[W + 1];
    float v = fmaxf(fmaxf(v0, v1), fmaxf(v2, v3));
    Out[i] = to_bf16(v);
}

// ---------------------------------------------------------------------------
extern "C" void kernel_launch(void* const* d_in, const int* in_sizes, int n_in,
                              void* d_out, int out_size, void* d_ws, size_t ws_size,
                              hipStream_t stream) {
    const float* x       = (const float*)d_in[0];
    const float* g_w     = (const float*)d_in[1];
    const float* g_b     = (const float*)d_in[2];
    const float* theta_w = (const float*)d_in[3];
    const float* theta_b = (const float*)d_in[4];
    const float* phi_w   = (const float*)d_in[5];
    const float* phi_b   = (const float*)d_in[6];
    const float* w_w     = (const float*)d_in[7];
    const float* w_b     = (const float*)d_in[8];
    const float* bn_g    = (const float*)d_in[9];
    const float* bn_b    = (const float*)d_in[10];
    const float* bn_m    = (const float*)d_in[11];
    const float* bn_v    = (const float*)d_in[12];
    float* out = (float*)d_out;

    // workspace layout (all bf16 buffers)
    char* ws = (char*)d_ws;
    size_t off = 0;
    auto alloc = [&](size_t bytes) { void* p = ws + off; off += (bytes + 255) & ~(size_t)255; return p; };
    bf16* xT      = (bf16*)alloc((size_t)Bn * N * C  * 2);   // (B,N,C)
    bf16* g_full  = (bf16*)alloc((size_t)Bn * CI * N * 2);   // (B,CI,N)
    bf16* phi_full= (bf16*)alloc((size_t)Bn * CI * N * 2);   // (B,CI,N)
    bf16* thetaT  = (bf16*)alloc((size_t)Bn * N * CI * 2);   // (B,N,CI)
    bf16* g_p     = (bf16*)alloc((size_t)Bn * CI * Mp * 2);  // (B,CI,M)
    bf16* phi_p   = (bf16*)alloc((size_t)Bn * CI * Mp * 2);  // (B,CI,M)
    bf16* kvT     = (bf16*)alloc((size_t)Bn * CI * CI * 2);  // (B,CI_d,CI_c)
    bf16* yT      = (bf16*)alloc((size_t)Bn * N * CI * 2);   // (B,N,CI)
    bf16* gw_h    = (bf16*)alloc((size_t)CI * C * 2);
    bf16* tw_h    = (bf16*)alloc((size_t)CI * C * 2);
    bf16* pw_h    = (bf16*)alloc((size_t)CI * C * 2);
    bf16* ww_h    = (bf16*)alloc((size_t)C * CI * 2);
    (void)ws_size; (void)n_in; (void)in_sizes; (void)out_size;

    // 1) weight conversion fp32 -> bf16
    {
        int n = CI * C;
        dim3 g((n + 255) / 256);
        f32_to_bf16_kernel<<<g, 256, 0, stream>>>(g_w, gw_h, n);
        f32_to_bf16_kernel<<<g, 256, 0, stream>>>(theta_w, tw_h, n);
        f32_to_bf16_kernel<<<g, 256, 0, stream>>>(phi_w, pw_h, n);
        f32_to_bf16_kernel<<<g, 256, 0, stream>>>(w_w, ww_h, n);
    }

    // 2) transpose x -> xT bf16
    transpose_f32_to_bf16<<<dim3(N / 32, C / 32, Bn), dim3(32, 8), 0, stream>>>(x, xT);

    // 3) projections: P = W (CI x C) * xT(N x C)^T  per batch
    {
        dim3 grid(N / 128, CI / 128, Bn);
        // g: (CI x N) normal layout, + bias
        gemm_nt_bf16<<<grid, 128, 0, stream>>>(gw_h, xT, g_full, g_b, 1.0f, 0,
                                               CI, N, C, 0, (size_t)N * C, (size_t)CI * N);
        // phi: (CI x N) normal layout, + bias
        gemm_nt_bf16<<<grid, 128, 0, stream>>>(pw_h, xT, phi_full, phi_b, 1.0f, 0,
                                               CI, N, C, 0, (size_t)N * C, (size_t)CI * N);
        // theta: transposed store -> thetaT (N x CI), + bias
        gemm_nt_bf16<<<grid, 128, 0, stream>>>(tw_h, xT, thetaT, theta_b, 1.0f, 1,
                                               CI, N, C, 0, (size_t)N * C, (size_t)N * CI);
    }

    // 4) max pool g/phi (1,2,2)
    {
        size_t total = (size_t)Bn * CI * Mp;
        dim3 g((unsigned)(total / 256));
        pool_hw<<<g, 256, 0, stream>>>(g_full, g_p);
        pool_hw<<<g, 256, 0, stream>>>(phi_full, phi_p);
    }

    // 5) kv[c,d] = (1/M) * sum_m phi_p[c,m] g_p[d,m]; store transposed -> kvT[d,c]
    gemm_nt_bf16<<<dim3(CI / 128, CI / 128, Bn), 128, 0, stream>>>(
        phi_p, g_p, kvT, nullptr, 1.0f / (float)Mp, 1,
        CI, CI, Mp, (size_t)CI * Mp, (size_t)CI * Mp, (size_t)CI * CI);

    // 6) y[d,n] = sum_c kvT[d,c] thetaT[n,c]; store transposed -> yT[n,d]
    gemm_nt_bf16<<<dim3(N / 128, CI / 128, Bn), 128, 0, stream>>>(
        kvT, thetaT, yT, nullptr, 1.0f, 1,
        CI, N, CI, (size_t)CI * CI, (size_t)N * CI, (size_t)N * CI);

    // 7) final: w_w * yT^T + w_b, BN, + x -> out (fp32)
    gemm_final<<<dim3(N / 128, C / 128, Bn), 128, 0, stream>>>(
        ww_h, yT, w_b, bn_g, bn_b, bn_m, bn_v, x, out,
        C, N, CI, (size_t)N * CI, (size_t)C * N);
}
